// MoE_44169443672206
// MI455X (gfx1250) — matrix operand order repaired and checked by reading, not compiled, not run on hardware
//
#include <hip/hip_runtime.h>

#define BB   2
#define LL   2048
#define DD   1024
#define EE   8
#define KK   2
#define DFF  4096
#define TT   (BB * LL)          // 4096 tokens

#define AM   128                // block M tile (rows)
#define TN   64                 // block N tile (cols)
#define TKS  32                 // K step (bf16 WMMA K=32)

typedef __attribute__((ext_vector_type(16))) __bf16 v16bf;
typedef __attribute__((ext_vector_type(2)))  __bf16 v2bf;
typedef __attribute__((ext_vector_type(8)))  float  v8f;

#if defined(__has_builtin)
#if __has_builtin(__builtin_amdgcn_cvt_pk_bf16_f32)
#define HAVE_CVT_PK_BF16 1
#endif
#endif

__device__ __forceinline__ unsigned pk2bf(float lo, float hi) {
#ifdef HAVE_CVT_PK_BF16
    v2bf v = __builtin_amdgcn_cvt_pk_bf16_f32(lo, hi);
    return __builtin_bit_cast(unsigned, v);
#else
    __bf16 l = (__bf16)lo, h = (__bf16)hi;
    return (unsigned)__builtin_bit_cast(unsigned short, l) |
           ((unsigned)__builtin_bit_cast(unsigned short, h) << 16);
#endif
}
__device__ __forceinline__ unsigned short f2bf(float f) {
    return __builtin_bit_cast(unsigned short, (__bf16)f);
}

// load 8 consecutive fp32, convert to 8 bf16 packed in a uint4
__device__ __forceinline__ uint4 ld8cvt(const float* __restrict__ p) {
    float4 a = ((const float4*)p)[0];
    float4 b = ((const float4*)p)[1];
    uint4 r;
    r.x = pk2bf(a.x, a.y);
    r.y = pk2bf(a.z, a.w);
    r.z = pk2bf(b.x, b.y);
    r.w = pk2bf(b.z, b.w);
    return r;
}

// A fragment (16x32 bf16): lane L<16 -> M=L, K=kofs..kofs+15 contiguous
__device__ __forceinline__ v16bf fragA(const unsigned short* __restrict__ row, int kofs) {
    v16bf f;
    ((uint4*)&f)[0] = *(const uint4*)(row + kofs);
    ((uint4*)&f)[1] = *(const uint4*)(row + kofs + 8);
    return f;
}
// B fragment (32x16 bf16), weights staged transposed in LDS as [n][k]:
// lane L<16 -> N=L, K=kb..kb+15 contiguous
__device__ __forceinline__ v16bf fragB(const unsigned short* __restrict__ row, int kb) {
    v16bf f;
    ((uint4*)&f)[0] = *(const uint4*)(row + kb);
    ((uint4*)&f)[1] = *(const uint4*)(row + kb + 8);
    return f;
}

__device__ __forceinline__ v8f wmma_bf16(v16bf a, v16bf b, v8f c) {
    return __builtin_amdgcn_wmma_f32_16x16x32_bf16(false, a, false, b, (short)0, c,
                                                   false, false);
}

// ---------------------------------------------------------------- router ----
__global__ __launch_bounds__(256) void moe_router(
    const float* __restrict__ x, const float* __restrict__ Wr,
    int* __restrict__ counts, int* __restrict__ tok,
    int* __restrict__ eids, int* __restrict__ slots, float* __restrict__ kw) {
    const int t    = (blockIdx.x * blockDim.x + threadIdx.x) >> 5;
    const int lane = threadIdx.x & 31;
    if (t >= TT) return;

    float acc[EE];
#pragma unroll
    for (int e = 0; e < EE; ++e) acc[e] = 0.f;
    for (int d = lane; d < DD; d += 32) {
        float xv = x[(size_t)t * DD + d];
#pragma unroll
        for (int e = 0; e < EE; ++e) acc[e] += xv * Wr[(size_t)e * DD + d];
    }
#pragma unroll
    for (int e = 0; e < EE; ++e) {
#pragma unroll
        for (int off = 16; off > 0; off >>= 1)
            acc[e] += __shfl_xor(acc[e], off, 32);
    }
    if (lane == 0) {
        float mx = acc[0];
#pragma unroll
        for (int e = 1; e < EE; ++e) mx = fmaxf(mx, acc[e]);
        float s[EE], sum = 0.f;
#pragma unroll
        for (int e = 0; e < EE; ++e) { s[e] = __expf(acc[e] - mx); sum += s[e]; }
        float inv = 1.f / sum;
#pragma unroll
        for (int e = 0; e < EE; ++e) s[e] *= inv;
        int i1 = 0; float v1 = s[0];
#pragma unroll
        for (int e = 1; e < EE; ++e) if (s[e] > v1) { v1 = s[e]; i1 = e; }
        int i2 = -1; float v2 = -1.f;
#pragma unroll
        for (int e = 0; e < EE; ++e) if (e != i1 && s[e] > v2) { v2 = s[e]; i2 = e; }
        float denom = v1 + v2 + 1.1920929e-7f;
        float w1 = v1 / denom, w2 = v2 / denom;

        int p1 = atomicAdd(&counts[i1], 1);
        tok[i1 * TT + p1] = t;
        eids[2 * t] = i1; slots[2 * t] = p1; kw[2 * t] = w1;
        int p2 = atomicAdd(&counts[i2], 1);
        tok[i2 * TT + p2] = t;
        eids[2 * t + 1] = i2; slots[2 * t + 1] = p2; kw[2 * t + 1] = w2;
    }
}

__global__ void moe_offsets(const int* __restrict__ counts, int* __restrict__ offs) {
    if (threadIdx.x == 0) {
        int a = 0;
#pragma unroll
        for (int e = 0; e < EE; ++e) { offs[e] = a; a += counts[e]; }
    }
}

// ------------------------------------------------------ gate+up (stage A) ----
// H[offs[e]+p, :] = relu(x[tok] @ Wg[e]^T) * (x[tok] @ Wu[e]^T), bf16
// block tile 128x64, wave tile 32x32, 8 WMMA per K step
__global__ __launch_bounds__(256) void moe_gateup(
    const float* __restrict__ x, const float* __restrict__ Wg,
    const float* __restrict__ Wu,
    const int* __restrict__ counts, const int* __restrict__ offs,
    const int* __restrict__ tok, unsigned short* __restrict__ H) {
    const int e  = blockIdx.z;
    const int m0 = blockIdx.y * AM;
    const int n0 = blockIdx.x * TN;
    const int ne = counts[e];
    if (m0 >= ne) return;
    const int hbase = offs[e];

    __shared__ __align__(16) unsigned short sA[2][AM][TKS];   // 16 KB
    __shared__ __align__(16) unsigned short sG[2][TN][TKS];   //  8 KB
    __shared__ __align__(16) unsigned short sU[2][TN][TKS];   //  8 KB
    __shared__ int stok[AM];

    const int tid = threadIdx.x;
    if (tid < AM) {
        int r = m0 + tid;
        stok[tid] = tok[e * TT + (r < ne ? r : 0)];
    }
    __syncthreads();

    // loader mapping: A rows by 2 threads (16 cols each), W rows by 4 threads
    const int arow = tid >> 1;            // 0..127
    const int acol = (tid & 1) << 4;      // 0,16
    const int lr = tid >> 2;              // 0..63
    const int lc = (tid & 3) << 3;        // 0,8,16,24

    const int lane = tid & 31;
    const int wv = tid >> 5;
    const int mbase = (wv & 3) * 32;      // wave 32-row footprint
    const int ni = (wv >> 2) * 32;        // wave 32-col footprint
    const int nlo = lane & 15;
    const int kofs = (lane >> 4) * 16;
    const int kb   = (lane >> 4) * 16;

    const size_t xrow = (size_t)stok[arow] * DD;
    const size_t wrow = ((size_t)e * DFF + n0 + lr) * DD;

    auto stage = [&](int b, int kk) {
        const float* xp = &x[xrow + kk + acol];
        *(uint4*)&sA[b][arow][acol]     = ld8cvt(xp);
        *(uint4*)&sA[b][arow][acol + 8] = ld8cvt(xp + 8);
        *(uint4*)&sG[b][lr][lc] = ld8cvt(&Wg[wrow + kk + lc]);
        *(uint4*)&sU[b][lr][lc] = ld8cvt(&Wu[wrow + kk + lc]);
    };

    v8f aG00 = {}, aG01 = {}, aG10 = {}, aG11 = {};
    v8f aU00 = {}, aU01 = {}, aU10 = {}, aU11 = {};

    const int NK = DD / TKS;              // 32
    stage(0, 0);
    for (int it = 0; it < NK; ++it) {
        __syncthreads();                  // buf (it&1) visible; prev reads retired
        if (it + 1 < NK) stage((it + 1) & 1, (it + 1) * TKS);
        const int b = it & 1;
        v16bf af0 = fragA(&sA[b][mbase + nlo][0], kofs);
        v16bf af1 = fragA(&sA[b][mbase + 16 + nlo][0], kofs);
        v16bf bg0 = fragB(&sG[b][ni + nlo][0], kb);
        v16bf bg1 = fragB(&sG[b][ni + 16 + nlo][0], kb);
        v16bf bu0 = fragB(&sU[b][ni + nlo][0], kb);
        v16bf bu1 = fragB(&sU[b][ni + 16 + nlo][0], kb);
        aG00 = wmma_bf16(af0, bg0, aG00);
        aG01 = wmma_bf16(af0, bg1, aG01);
        aG10 = wmma_bf16(af1, bg0, aG10);
        aG11 = wmma_bf16(af1, bg1, aG11);
        aU00 = wmma_bf16(af0, bu0, aU00);
        aU01 = wmma_bf16(af0, bu1, aU01);
        aU10 = wmma_bf16(af1, bu0, aU10);
        aU11 = wmma_bf16(af1, bu1, aU11);
    }

    const int rsub = (lane >> 4) * 8;
    v8f G[2][2] = {{aG00, aG01}, {aG10, aG11}};
    v8f U[2][2] = {{aU00, aU01}, {aU10, aU11}};
#pragma unroll
    for (int mf = 0; mf < 2; ++mf) {
#pragma unroll
        for (int i = 0; i < 8; ++i) {
            int row = mbase + mf * 16 + rsub + i;
            if (m0 + row < ne) {
                size_t hr = (size_t)(hbase + m0 + row) * DFF;
                float h0 = fmaxf(G[mf][0][i], 0.f) * U[mf][0][i];
                float h1 = fmaxf(G[mf][1][i], 0.f) * U[mf][1][i];
                H[hr + n0 + ni + nlo]      = f2bf(h0);
                H[hr + n0 + ni + 16 + nlo] = f2bf(h1);
            }
        }
    }
}

// --------------------------------------------------------- down (stage B) ----
// P[offs[e]+p, :] = H[offs[e]+p, :] @ Wd[e]^T  (unweighted partials, fp32)
// block tile 128x64, wave tile 32x32, 4 WMMA per K step
__global__ __launch_bounds__(256) void moe_down(
    const unsigned short* __restrict__ H, const float* __restrict__ Wd,
    const int* __restrict__ counts, const int* __restrict__ offs,
    float* __restrict__ P) {
    const int e  = blockIdx.z;
    const int m0 = blockIdx.y * AM;
    const int n0 = blockIdx.x * TN;
    const int ne = counts[e];
    if (m0 >= ne) return;
    const int hbase = offs[e];

    __shared__ __align__(16) unsigned short sH[2][AM][TKS];   // 16 KB
    __shared__ __align__(16) unsigned short sW[2][TN][TKS];   //  8 KB

    const int tid = threadIdx.x;
    const int arow = tid >> 1;
    const int acol = (tid & 1) << 4;
    const int lr = tid >> 2;
    const int lc = (tid & 3) << 3;
    const int lane = tid & 31;
    const int wv = tid >> 5;
    const int mbase = (wv & 3) * 32;
    const int ni = (wv >> 2) * 32;
    const int nlo = lane & 15;
    const int kofs = (lane >> 4) * 16;
    const int kb   = (lane >> 4) * 16;

    const size_t hrow = (size_t)(hbase + m0 + arow) * DFF;
    const size_t wrow = ((size_t)e * DD + n0 + lr) * DFF;

    auto stage = [&](int b, int kk) {
        const unsigned short* hp = &H[hrow + kk + acol];      // already bf16
        *(uint4*)&sH[b][arow][acol]     = ((const uint4*)hp)[0];
        *(uint4*)&sH[b][arow][acol + 8] = ((const uint4*)hp)[1];
        *(uint4*)&sW[b][lr][lc] = ld8cvt(&Wd[wrow + kk + lc]);
    };

    v8f a00 = {}, a01 = {}, a10 = {}, a11 = {};

    const int NK = DFF / TKS;             // 128
    stage(0, 0);
    for (int it = 0; it < NK; ++it) {
        __syncthreads();
        if (it + 1 < NK) stage((it + 1) & 1, (it + 1) * TKS);
        const int b = it & 1;
        v16bf af0 = fragA(&sH[b][mbase + nlo][0], kofs);
        v16bf af1 = fragA(&sH[b][mbase + 16 + nlo][0], kofs);
        v16bf b0  = fragB(&sW[b][ni + nlo][0], kb);
        v16bf b1  = fragB(&sW[b][ni + 16 + nlo][0], kb);
        a00 = wmma_bf16(af0, b0, a00);
        a01 = wmma_bf16(af0, b1, a01);
        a10 = wmma_bf16(af1, b0, a10);
        a11 = wmma_bf16(af1, b1, a11);
    }

    const int rsub = (lane >> 4) * 8;
    v8f A[2][2] = {{a00, a01}, {a10, a11}};
#pragma unroll
    for (int mf = 0; mf < 2; ++mf) {
#pragma unroll
        for (int i = 0; i < 8; ++i) {
            int row = mbase + mf * 16 + rsub + i;
            if (m0 + row < ne) {
                size_t pr = (size_t)(hbase + m0 + row) * DD;
                P[pr + n0 + ni + nlo]      = A[mf][0][i];
                P[pr + n0 + ni + 16 + nlo] = A[mf][1][i];
            }
        }
    }
}

// ------------------------------------------------------------- combine ------
// out[t] = w1 * P[offs[e1]+slot1] + w2 * P[offs[e2]+slot2]  (fixed k order)
__global__ __launch_bounds__(256) void moe_combine(
    const float* __restrict__ P, const int* __restrict__ offs,
    const int* __restrict__ eids, const int* __restrict__ slots,
    const float* __restrict__ kw, float* __restrict__ out) {
    int gid = blockIdx.x * blockDim.x + threadIdx.x;   // TT * (DD/4) threads
    int t  = gid >> 8;                                 // DD/4 == 256
    int c4 = (gid & 255) << 2;
    int e1 = eids[2 * t], e2 = eids[2 * t + 1];
    size_t r1 = (size_t)(offs[e1] + slots[2 * t]) * DD;
    size_t r2 = (size_t)(offs[e2] + slots[2 * t + 1]) * DD;
    float w1 = kw[2 * t], w2 = kw[2 * t + 1];
    float4 a = *(const float4*)&P[r1 + c4];
    float4 b = *(const float4*)&P[r2 + c4];
    float4 o;
    o.x = w1 * a.x + w2 * b.x;
    o.y = w1 * a.y + w2 * b.y;
    o.z = w1 * a.z + w2 * b.z;
    o.w = w1 * a.w + w2 * b.w;
    *(float4*)&out[(size_t)t * DD + c4] = o;
}

// ---------------------------------------------------------------------------
extern "C" void kernel_launch(void* const* d_in, const int* in_sizes, int n_in,
                              void* d_out, int out_size, void* d_ws, size_t ws_size,
                              hipStream_t stream) {
    (void)in_sizes; (void)n_in; (void)out_size; (void)ws_size;
    const float* x  = (const float*)d_in[0];
    const float* Wr = (const float*)d_in[1];
    const float* Wg = (const float*)d_in[2];
    const float* Wu = (const float*)d_in[3];
    const float* Wd = (const float*)d_in[4];
    float* out = (float*)d_out;

    char* ws = (char*)d_ws;
    // workspace layout (bytes)
    int*   counts = (int*)(ws + 0);                       // 8 ints
    int*   offs   = (int*)(ws + 64);                      // 8 ints
    int*   tok    = (int*)(ws + 256);                     // EE*TT ints   = 131072 B
    int*   eids   = (int*)(ws + 256 + 131072);            // 2*TT ints    = 32768 B
    int*   slots  = (int*)(ws + 256 + 131072 + 32768);    // 2*TT ints    = 32768 B
    float* kw     = (float*)(ws + 256 + 131072 + 65536);  // 2*TT floats  = 32768 B
    const size_t HOFF = 229632;                           // 256-aligned
    const size_t ROWS = (size_t)TT * KK + AM;             // slack rows for tile reads
    unsigned short* H = (unsigned short*)(ws + HOFF);     // ROWS*DFF bf16 = ~68 MB
    const size_t POFF = HOFF + ROWS * DFF * 2;
    float* P = (float*)(ws + POFF);                       // ROWS*DD fp32  = ~34 MB

    hipMemsetAsync(counts, 0, 64, stream);

    moe_router<<<TT / 8, 256, 0, stream>>>(x, Wr, counts, tok, eids, slots, kw);
    moe_offsets<<<1, 32, 0, stream>>>(counts, offs);

    dim3 gA(DFF / TN, TT / AM, EE);   // 64 x 32 x 8
    moe_gateup<<<gA, 256, 0, stream>>>(x, Wg, Wu, counts, offs, tok, H);

    dim3 gB(DD / TN, TT / AM, EE);    // 16 x 32 x 8
    moe_down<<<gB, 256, 0, stream>>>(H, Wd, counts, offs, P);

    moe_combine<<<(TT * (DD / 4)) / 256, 256, 0, stream>>>(P, offs, eids, slots, kw, out);
}
